// KDGCN_2886218022958
// MI455X (gfx1250) — matrix operand order repaired and checked by reading, not compiled, not run on hardware
//
#include <hip/hip_runtime.h>
#include <hip/hip_bf16.h>

typedef __attribute__((ext_vector_type(16))) __bf16 v16bf;
typedef __attribute__((ext_vector_type(4)))  __bf16 v4bf;
typedef __attribute__((ext_vector_type(8)))  float  v8f;

#define EPSV 1e-5f

__device__ __forceinline__ __bf16 f2bf(float f) {
    unsigned u = __builtin_bit_cast(unsigned, f);
    u += 0x7FFFu + ((u >> 16) & 1u);              // round-to-nearest-even
    unsigned short s = (unsigned short)(u >> 16);
    return __builtin_bit_cast(__bf16, s);
}

// ---------------------------------------------------------------------------
// GEMM: C[n][CO] = A[n][128] * W[128][CO], bf16 WMMA, fp32 accumulate.
// Block = 256 threads = 8 waves; block owns 128 rows; wave w owns rows
// [16w,16w+16) x all CO cols. LDS tiles are stored in WMMA *fragment order*
// ([tile][kstep][lane][e]) so each lane fetches its whole 32B fragment with
// one aligned vector load (2x ds_load_b128), no per-element gather/packing.
//
// 16-bit A (16x32) layout inverse:  lane = mRow + 16*bit3(kin),
//                                   e    = (kin&7) | ((kin&16)>>1)
// 16-bit B (32x16) layout inverse:  lane = colN + 16*bit3(kin),
//                                   e    = (kin&7) | ((kin&16)>>1)
// ---------------------------------------------------------------------------
template <int CO>
__global__ __launch_bounds__(256) void gcn_gemm_bf16(
    const float* __restrict__ A, const float* __restrict__ W,
    float* __restrict__ C, int nrows)
{
    constexpr int NT = CO / 16;                    // column tiles
    __shared__ __align__(32) __bf16 sA[8 * 4 * 32 * 16];      // 32 KB
    __shared__ __align__(32) __bf16 sW[4 * NT * 32 * 16];     // 32/16 KB

    const int tid  = threadIdx.x;
    const int lane = tid & 31;
    const int wave = tid >> 5;
    const long long rowBase = (long long)blockIdx.x * 128;

    // ---- stage W into fragment order (once per block) ----
    for (int i = tid; i < 128 * CO; i += 256) {
        int k = i / CO, co = i % CO;               // W row-major [128][CO]
        int kstep = k >> 5, kin = k & 31;
        int bl = (co & 15) + (((kin >> 3) & 1) << 4); // lane
        int be = (kin & 7) | ((kin & 16) >> 1);    // element
        sW[(((kstep * NT + (co >> 4)) << 5) + bl) * 16 + be] = f2bf(W[i]);
    }

    // ---- stage A tile into fragment order: float4 loads -> v4bf LDS store ----
    for (int i4 = tid; i4 < 128 * 32; i4 += 256) { // i4 indexes a float4
        int r  = i4 >> 5;                          // tile row 0..127
        int c4 = (i4 & 31) << 2;                   // col 0,4,...,124
        long long gr = rowBase + r;
        float4 v = make_float4(0.f, 0.f, 0.f, 0.f);
        if (gr < nrows) v = *reinterpret_cast<const float4*>(A + gr * 128 + c4);
        int tileM = r >> 4, mRow = r & 15;
        int kstep = c4 >> 5, kin = c4 & 31;
        int al = mRow + (((kin >> 3) & 1) << 4);   // lane
        int ae = (kin & 7) | ((kin & 16) >> 1);    // element (multiple of 4 here)
        v4bf pk = { f2bf(v.x), f2bf(v.y), f2bf(v.z), f2bf(v.w) };
        *reinterpret_cast<v4bf*>(&sA[((((tileM << 2) + kstep) << 5) + al) * 16 + ae]) = pk;
    }
    __syncthreads();

    v8f acc[NT];
#pragma unroll
    for (int t = 0; t < NT; ++t)
        acc[t] = (v8f){0.f,0.f,0.f,0.f,0.f,0.f,0.f,0.f};

    const __bf16* aBase = &sA[((wave << 2) * 32 + lane) * 16];
    const __bf16* bBase = &sW[lane * 16];

#pragma unroll
    for (int kk = 0; kk < 4; ++kk) {
        v16bf afrag = *reinterpret_cast<const v16bf*>(aBase + (kk << 5) * 16);
#pragma unroll
        for (int t = 0; t < NT; ++t) {
            v16bf bfrag = *reinterpret_cast<const v16bf*>(bBase + (((kk * NT + t) << 5)) * 16);
            acc[t] = __builtin_amdgcn_wmma_f32_16x16x32_bf16(
                false, afrag, false, bfrag, (short)0, acc[t], false, false);
        }
    }

    // D layout: lane l, vgpr v -> row = v + (l>=16 ? 8 : 0), col = l&15
    const int mRow = lane & 15;
    const int rOff = (lane >= 16) ? 8 : 0;
#pragma unroll
    for (int t = 0; t < NT; ++t) {
#pragma unroll
        for (int v = 0; v < 8; ++v) {
            long long gr = rowBase + wave * 16 + v + rOff;
            if (gr < nrows) C[gr * CO + t * 16 + mRow] = acc[t][v];
        }
    }
}

// ---------------------------------------------------------------------------
// Degree / dinv
// ---------------------------------------------------------------------------
__global__ void k_deg_init(float* deg, int n) {
    long long i = (long long)blockIdx.x * blockDim.x + threadIdx.x;
    if (i < n) deg[i] = 1.0f;                      // self loop
}
__global__ void k_deg_edges(const long long* __restrict__ dst, float* deg, long long nE) {
    long long i = (long long)blockIdx.x * blockDim.x + threadIdx.x;
    if (i < nE) atomicAdd(&deg[dst[i]], 1.0f);
}
__global__ void k_dinv(const float* deg, float* dinv, int n) {
    long long i = (long long)blockIdx.x * blockDim.x + threadIdx.x;
    if (i < n) dinv[i] = rsqrtf(fmaxf(deg[i], 1.0f));
}

// ---------------------------------------------------------------------------
// Aggregation: out[n][c] = b[c] + h[n][c]*dinv[n]^2  (self loop + bias init),
// then edge scatter: out[dst] += h[src] * dinv[src]*dinv[dst]  (L2 atomics;
// the 51 MB feature buffer is L2-resident on the 192 MB L2).
// ---------------------------------------------------------------------------
template <int C>
__global__ void k_agg_init(const float* __restrict__ h, const float* __restrict__ bias,
                           const float* __restrict__ dinv, float* __restrict__ out, int n)
{
    long long i = (long long)blockIdx.x * blockDim.x + threadIdx.x;
    long long total = (long long)n * C;
    if (i >= total) return;
    int node = (int)(i / C), c = (int)(i % C);
    float d = dinv[node];
    out[i] = bias[c] + h[i] * d * d;
}

template <int C>
__global__ __launch_bounds__(256) void k_agg_edges(
    const float* __restrict__ h, const long long* __restrict__ src,
    const long long* __restrict__ dst, const float* __restrict__ dinv,
    float* __restrict__ out, long long nE)
{
    const int lane = threadIdx.x & 31;
    long long e = (long long)blockIdx.x * 8 + (threadIdx.x >> 5); // 1 wave / edge
    if (e >= nE) return;
    long long s = src[e], d = dst[e];
    float norm = dinv[s] * dinv[d];
    constexpr int V = C / 32;                      // floats per lane (4 or 2)
    const float* hp = h + s * C + lane * V;
    float* op = out + d * C + lane * V;
#pragma unroll
    for (int j = 0; j < V; ++j) atomicAdd(op + j, hp[j] * norm);
}

// ---------------------------------------------------------------------------
// BatchNorm (training stats) + ReLU
// ---------------------------------------------------------------------------
__global__ void k_stats_zero(float* stats) {
    if (threadIdx.x < 256) stats[threadIdx.x] = 0.0f;
}
template <int C>
__global__ __launch_bounds__(C) void k_bn_stats(const float* __restrict__ h,
                                                float* __restrict__ stats, int n)
{
    const int c = threadIdx.x;                     // one thread per channel
    long long r0 = (long long)blockIdx.x * 128;
    float s = 0.f, s2 = 0.f;
    for (int r = 0; r < 128; ++r) {
        long long row = r0 + r;
        if (row >= n) break;
        float v = h[row * C + c];
        s += v; s2 += v * v;
    }
    atomicAdd(&stats[c], s);
    atomicAdd(&stats[C + c], s2);
}
template <int C>
__global__ void k_bn_apply(const float* __restrict__ h, const float* __restrict__ stats,
                           const float* __restrict__ gamma, const float* __restrict__ beta,
                           float* __restrict__ out, int n)
{
    long long i = (long long)blockIdx.x * blockDim.x + threadIdx.x;
    long long total = (long long)n * C;
    if (i >= total) return;
    int c = (int)(i % C);
    float inv_n = 1.0f / (float)n;
    float mean = stats[c] * inv_n;
    float var  = stats[C + c] * inv_n - mean * mean;
    float y = gamma[c] * (h[i] - mean) * rsqrtf(var + EPSV) + beta[c];
    out[i] = fmaxf(y, 0.0f);
}

// ---------------------------------------------------------------------------
// Host launcher
// ---------------------------------------------------------------------------
extern "C" void kernel_launch(void* const* d_in, const int* in_sizes, int n_in,
                              void* d_out, int out_size, void* d_ws, size_t ws_size,
                              hipStream_t stream)
{
    const float* x  = (const float*)d_in[0];
    const float* W1 = (const float*)d_in[1];
    const float* b1 = (const float*)d_in[2];
    const float* g1 = (const float*)d_in[3];
    const float* be1= (const float*)d_in[4];
    const float* W2 = (const float*)d_in[5];
    const float* b2 = (const float*)d_in[6];
    const float* g2 = (const float*)d_in[7];
    const float* be2= (const float*)d_in[8];
    const float* W3 = (const float*)d_in[9];
    const float* b3 = (const float*)d_in[10];
    const long long* ei = (const long long*)d_in[11];

    const int  N = in_sizes[0] / 128;
    const long long E = (long long)in_sizes[11] / 2;
    const long long* src = ei;
    const long long* dst = ei + E;

    // workspace (floats): deg[N] | dinv[N] | bufA[N*128] | bufB[N*128] | stats[256]
    float* deg   = (float*)d_ws;
    float* dinv  = deg + N;
    float* bufA  = dinv + N;
    float* bufB  = bufA + (long long)N * 128;
    float* stats = bufB + (long long)N * 128;

    float* h_out = (float*)d_out;                      // N x 128
    float* out3  = (float*)d_out + (long long)N * 128; // N x 64

    const int B256_N   = (N + 255) / 256;
    const int B256_E   = (int)((E + 255) / 256);
    const int B_GEMM   = (N + 127) / 128;
    const int B_EDGE   = (int)((E + 7) / 8);
    const int B_ELT128 = (int)(((long long)N * 128 + 255) / 256);
    const int B_ELT64  = (int)(((long long)N * 64 + 255) / 256);
    const int B_STATS  = (N + 127) / 128;

    // degrees (with self loops) and D^{-1/2}
    k_deg_init <<<B256_N, 256, 0, stream>>>(deg, N);
    k_deg_edges<<<B256_E, 256, 0, stream>>>(dst, deg, E);
    k_dinv     <<<B256_N, 256, 0, stream>>>(deg, dinv, N);

    // ---- Layer 1: x @ W1 -> aggregate -> BN+ReLU ----
    gcn_gemm_bf16<128><<<B_GEMM, 256, 0, stream>>>(x, W1, bufA, N);
    k_agg_init <128>  <<<B_ELT128, 256, 0, stream>>>(bufA, b1, dinv, bufB, N);
    k_agg_edges<128>  <<<B_EDGE, 256, 0, stream>>>(bufA, src, dst, dinv, bufB, E);
    k_stats_zero      <<<1, 256, 0, stream>>>(stats);
    k_bn_stats <128>  <<<B_STATS, 128, 0, stream>>>(bufB, stats, N);
    k_bn_apply <128>  <<<B_ELT128, 256, 0, stream>>>(bufB, stats, g1, be1, bufA, N);

    // ---- Layer 2 ----
    gcn_gemm_bf16<128><<<B_GEMM, 256, 0, stream>>>(bufA, W2, bufB, N);
    k_agg_init <128>  <<<B_ELT128, 256, 0, stream>>>(bufB, b2, dinv, bufA, N);
    k_agg_edges<128>  <<<B_EDGE, 256, 0, stream>>>(bufB, src, dst, dinv, bufA, E);
    k_stats_zero      <<<1, 256, 0, stream>>>(stats);
    k_bn_stats <128>  <<<B_STATS, 128, 0, stream>>>(bufA, stats, N);
    k_bn_apply <128>  <<<B_ELT128, 256, 0, stream>>>(bufA, stats, g2, be2, h_out, N);

    // ---- Layer 3: h_out @ W3 -> aggregate into d_out tail ----
    gcn_gemm_bf16<64> <<<B_GEMM, 256, 0, stream>>>(h_out, W3, bufB, N); // bufB: N x 64
    k_agg_init <64>   <<<B_ELT64, 256, 0, stream>>>(bufB, b3, dinv, out3, N);
    k_agg_edges<64>   <<<B_EDGE, 256, 0, stream>>>(bufB, src, dst, dinv, out3, E);
}
